// SeNaTraBlock_29111288332437
// MI455X (gfx1250) — matrix operand order repaired and checked by reading, not compiled, not run on hardware
//
#include <hip/hip_runtime.h>
#include <hip/hip_bf16.h>
#include <math.h>

// Problem constants
#define BD     4
#define HH     56
#define WW     56
#define NN     3136          // H*W
#define DIMC   128
#define HEADS  4
#define HDIM   32
#define HIDDEN 384
#define MTOT   (BD*NN)       // 12544 rows
#define SCALE  0.17677669529663687f   // 32^-0.5
#define KSPAN  22            // max key-column span for a 16-query tile
#define KSLOTS 160           // 10 tiles of 16 (154 real key slots, padded)

typedef __attribute__((ext_vector_type(16))) __bf16 v16bf;
typedef __attribute__((ext_vector_type(8)))  float  v8f;
typedef unsigned int __attribute__((may_alias)) u32a;

union FB { v16bf v; unsigned u[8]; };
union F8 { v8f  v; float    f[8]; };

__device__ __forceinline__ unsigned short f2bf(float x){
  unsigned u = __float_as_uint(x);
  u += 0x7fffu + ((u >> 16) & 1u);           // round-to-nearest-even
  return (unsigned short)(u >> 16);
}

// A fragment (16x32 bf16, MxK): lane m = lane&15; hi=lane>>4 selects K block of 16.
// VGPR i -> K = 16*hi + 2i; pairs of halves = one dword.
__device__ __forceinline__ v16bf load_a16(const unsigned short* base, int stride, int lane){
  FB f;
  const unsigned short* row = base + (lane & 15) * stride + (lane >> 4) * 16;
#pragma unroll
  for (int i = 0; i < 8; ++i){
    int kb = 2 * i;
    f.u[i] = *(const u32a*)(row + kb);
  }
  return f.v;
}

// B fragment (32x16 bf16, KxN) from an [N][K] row-major source (stride in halves):
// lane n = lane&15; hi=lane>>4 selects K block of 16; halves sequential.
__device__ __forceinline__ v16bf load_bnk(const unsigned short* base, int stride, int lane){
  FB f;
  const unsigned short* row = base + (lane & 15) * stride + (lane >> 4) * 16;
#pragma unroll
  for (int i = 0; i < 8; ++i) f.u[i] = *(const u32a*)(row + 2 * i);
  return f.v;
}

__device__ __forceinline__ v8f wmma_bf(v16bf a, v16bf b, v8f c){
  return __builtin_amdgcn_wmma_f32_16x16x32_bf16(false, a, false, b, (short)0, c, false, false);
}

__device__ __forceinline__ int iclip(int v, int lo, int hi){ return v < lo ? lo : (v > hi ? hi : v); }

// ---------------------------------------------------------------- weight cvt
__global__ void k_cvt(const float* __restrict__ s, unsigned short* __restrict__ d, int n){
  int i = blockIdx.x * 256 + threadIdx.x;
  if (i < n) d[i] = f2bf(s[i]);
}

// ------------------------------------------------- LN1 + QKV GEMM + RoPE
__global__ __launch_bounds__(128) void k_ln_qkv_rope(
    const float* __restrict__ x, const float* __restrict__ g1, const float* __restrict__ b1,
    const unsigned short* __restrict__ wqkv,
    unsigned short* __restrict__ qb, unsigned short* __restrict__ kb,
    unsigned short* __restrict__ vb)
{
  __shared__ unsigned short lnA[64 * DIMC];
  int lane = threadIdx.x & 31, wv = threadIdx.x >> 5;
  int ln = lane & 15, hi = lane >> 4;
  int g0 = blockIdx.x * 64 + wv * 16;

  // LayerNorm 16 rows (lane-parallel over 128 cols), write bf16 tile to LDS
  for (int rr = 0; rr < 16; ++rr){
    int g = g0 + rr;
    const float4* xr = (const float4*)(x + (size_t)g * DIMC);
    float4 v = xr[lane];
    float s  = v.x + v.y + v.z + v.w;
    float s2 = v.x*v.x + v.y*v.y + v.z*v.z + v.w*v.w;
#pragma unroll
    for (int d = 1; d < 32; d <<= 1){ s += __shfl_xor(s, d, 32); s2 += __shfl_xor(s2, d, 32); }
    float mu  = s  * (1.0f / DIMC);
    float var = s2 * (1.0f / DIMC) - mu * mu;
    float inv = rsqrtf(var + 1e-5f);
    int c0 = lane * 4;
    unsigned short* dst = lnA + (wv * 16 + rr) * DIMC + c0;
    dst[0] = f2bf((v.x - mu) * inv * g1[c0+0] + b1[c0+0]);
    dst[1] = f2bf((v.y - mu) * inv * g1[c0+1] + b1[c0+1]);
    dst[2] = f2bf((v.z - mu) * inv * g1[c0+2] + b1[c0+2]);
    dst[3] = f2bf((v.w - mu) * inv * g1[c0+3] + b1[c0+3]);
  }
  // Each wave reads only its own 16 rows -> no barrier needed.
  v16bf af[4];
#pragma unroll
  for (int ks = 0; ks < 4; ++ks) af[ks] = load_a16(lnA + (wv*16)*DIMC + ks*32, DIMC, lane);

  for (int ct = 0; ct < 24; ++ct){
    F8 c;
#pragma unroll
    for (int i = 0; i < 8; ++i) c.f[i] = 0.f;
#pragma unroll
    for (int ks = 0; ks < 4; ++ks){
      v16bf bf = load_bnk(wqkv + (ct*16)*DIMC + ks*32, DIMC, lane);
      c.v = wmma_bf(af[ks], bf, c.v);
    }
    int o    = ct*16 + ln;               // output channel 0..383
    int head = (o & 127) >> 5;
    int hd   = o & 31;
    float invf = __expf(-0.5756462732f * (float)(hd >> 1));   // 10000^(-2p/32)
    bool isq = (o < 128), isk = (o >= 128 && o < 256);
#pragma unroll
    for (int r = 0; r < 8; ++r){
      float val  = c.f[r];
      float part = __shfl_xor(val, 1, 32);   // RoPE partner (o^1)
      int m = r + 8*hi;
      int g = g0 + m;
      int b_ = g / NN, tok = g % NN;
      float outv = val;
      if (o < 256){
        float ang = (float)tok * invf;
        float sn, cs; __sincosf(ang, &sn, &cs);
        outv = (o & 1) ? (val*cs + part*sn) : (val*cs - part*sn);
      }
      size_t idx = ((size_t)(b_*HEADS + head)*NN + tok)*HDIM + hd;
      if (isq)      qb[idx] = f2bf(outv);
      else if (isk) kb[idx] = f2bf(outv);
      else          vb[idx] = f2bf(val);
    }
  }
}

// ------------------------------------------------- neighborhood attention
__global__ __launch_bounds__(128) void k_attn(
    const unsigned short* __restrict__ qb, const unsigned short* __restrict__ kb,
    const unsigned short* __restrict__ vb, unsigned short* __restrict__ attn)
{
  __shared__ unsigned short smem[4 * (32*KSLOTS + 16*KSLOTS)];   // per-wave V^T + P
  int lane = threadIdx.x & 31, wv = threadIdx.x >> 5;
  int ln = lane & 15, hi = lane >> 4;
  int i  = blockIdx.x % HH;
  int bh = blockIdx.x / HH;            // b*HEADS + h
  int j0 = (wv < 3) ? wv*16 : 40;      // last tile overlaps (recompute)
  int rstart = iclip(i - 3, 0, HH - 7);
  int cb = iclip(j0 - 3, 0, WW - 7);
  int ce = iclip(j0 + 15 - 3, 0, WW - 7) + 7;
  int cspan = ce - cb;                 // 19..22

  unsigned short* vT = smem + wv * (48*KSLOTS);   // [32][160] (channel-major)
  unsigned short* pm = vT + 32 * KSLOTS;          // [16][160] probabilities
  const unsigned short* kg = kb + (size_t)bh * NN * HDIM;
  const unsigned short* vg = vb + (size_t)bh * NN * HDIM;
  const unsigned short* qg = qb + ((size_t)bh * NN + i*WW + j0) * HDIM;

  v16bf qa = load_a16(qg, HDIM, lane);            // 16 queries x 32 channels

  // Stage V transposed into LDS: vT[ch][key], all 160 slots (clamped addresses)
  for (int t = lane; t < KSLOTS * 16; t += 32){
    int key = t >> 4, cp = (t & 15) * 2;
    int kc = key < 153 ? key : 153;
    int di = kc / KSPAN, dj = kc % KSPAN; if (dj > cspan - 1) dj = cspan - 1;
    unsigned vv = *(const u32a*)(vg + (((rstart + di) * WW) + (cb + dj)) * HDIM + cp);
    vT[ cp      * KSLOTS + key] = (unsigned short)(vv & 0xffffu);
    vT[(cp + 1) * KSLOTS + key] = (unsigned short)(vv >> 16);
  }

  // S = Q K^T : 10 key tiles of 16, B fragments loaded straight from global
  F8 acc[10];
  for (int kt = 0; kt < 10; ++kt){
    int key = kt*16 + ln;
    int kc = key < 153 ? key : 153;
    int di = kc / KSPAN, dj = kc % KSPAN; if (dj > cspan - 1) dj = cspan - 1;
    const unsigned short* row = kg + (((rstart + di) * WW) + (cb + dj)) * HDIM + hi * 16;
    FB f;
#pragma unroll
    for (int q2 = 0; q2 < 8; ++q2) f.u[q2] = *(const u32a*)(row + 2*q2);
    F8 z;
#pragma unroll
    for (int q2 = 0; q2 < 8; ++q2) z.f[q2] = 0.f;
    acc[kt].v = wmma_bf(qa, f.v, z.v);
  }

  // Masked softmax per query row (keys live across 16 lanes of each D fragment)
#pragma unroll
  for (int r = 0; r < 8; ++r){
    int m = r + 8*hi;
    int sq = iclip(j0 + m - 3, 0, WW - 7);
    float mx = -1e30f;
#pragma unroll
    for (int kt = 0; kt < 10; ++kt){
      int key = kt*16 + ln;
      int dj = key % KSPAN, cj = cb + dj;
      bool valid = (key < 154) && (dj < cspan) && (cj >= sq) && (cj < sq + 7);
      float s = valid ? acc[kt].f[r] * SCALE : -1e30f;
      acc[kt].f[r] = s;
      mx = fmaxf(mx, s);
    }
#pragma unroll
    for (int d = 1; d < 16; d <<= 1) mx = fmaxf(mx, __shfl_xor(mx, d, 32));
    float sum = 0.f;
#pragma unroll
    for (int kt = 0; kt < 10; ++kt){
      float p = __expf(acc[kt].f[r] - mx);
      acc[kt].f[r] = p; sum += p;
    }
#pragma unroll
    for (int d = 1; d < 16; d <<= 1) sum += __shfl_xor(sum, d, 32);
    float isum = 1.f / sum;
#pragma unroll
    for (int kt = 0; kt < 10; ++kt)
      pm[m * KSLOTS + kt*16 + ln] = f2bf(acc[kt].f[r] * isum);
  }

  // O = P V : K over 160 key slots (5 steps), two 16-wide channel tiles
  F8 o0, o1;
#pragma unroll
  for (int q2 = 0; q2 < 8; ++q2){ o0.f[q2] = 0.f; o1.f[q2] = 0.f; }
  for (int ks = 0; ks < 5; ++ks){
    v16bf pa  = load_a16(pm + ks*32, KSLOTS, lane);
    v16bf bv0 = load_bnk(vT + ks*32, KSLOTS, lane);
    v16bf bv1 = load_bnk(vT + 16*KSLOTS + ks*32, KSLOTS, lane);
    o0.v = wmma_bf(pa, bv0, o0.v);
    o1.v = wmma_bf(pa, bv1, o1.v);
  }
  // Exclusive write ownership (deterministic output):
  //   wave 0: j 0-15, wave 1: j 16-31, wave 2: j 32-39 (m<8 only), wave 3: j 40-55.
  int b_ = bh >> 2, h_ = bh & 3;
#pragma unroll
  for (int r = 0; r < 8; ++r){
    int m = r + 8*hi;
    bool own = (wv != 2) || (m < 8);
    if (own){
      int tok = i*WW + j0 + m;
      size_t base = ((size_t)b_*NN + tok) * DIMC + h_*HDIM;
      attn[base + ln]      = f2bf(o0.f[r]);
      attn[base + 16 + ln] = f2bf(o1.f[r]);
    }
  }
}

// ------------------------------------------------- proj GEMM + bias + residual
__global__ __launch_bounds__(128) void k_proj(
    const unsigned short* __restrict__ attn, const unsigned short* __restrict__ wproj,
    const float* __restrict__ pbias, const float* __restrict__ x, float* __restrict__ x2)
{
  int lane = threadIdx.x & 31, wv = threadIdx.x >> 5;
  int ln = lane & 15, hi = lane >> 4;
  int g0 = blockIdx.x * 64 + wv * 16;
  v16bf af[4];
#pragma unroll
  for (int ks = 0; ks < 4; ++ks) af[ks] = load_a16(attn + (size_t)g0*DIMC + ks*32, DIMC, lane);
  for (int ct = 0; ct < 8; ++ct){
    F8 c;
#pragma unroll
    for (int i = 0; i < 8; ++i) c.f[i] = 0.f;
#pragma unroll
    for (int ks = 0; ks < 4; ++ks)
      c.v = wmma_bf(af[ks], load_bnk(wproj + (ct*16)*DIMC + ks*32, DIMC, lane), c.v);
    int o = ct*16 + ln;
    float bo = pbias[o];
#pragma unroll
    for (int r = 0; r < 8; ++r){
      int g = g0 + r + 8*hi;
      x2[(size_t)g*DIMC + o] = c.f[r] + bo + x[(size_t)g*DIMC + o];
    }
  }
}

// ------------------------------------------------- LN2 + FC1 + exact GELU
__global__ __launch_bounds__(128) void k_ln_fc1(
    const float* __restrict__ x2, const float* __restrict__ g2, const float* __restrict__ b2v,
    const unsigned short* __restrict__ w1, const float* __restrict__ bias1,
    unsigned short* __restrict__ hbuf)
{
  __shared__ unsigned short lnA[64 * DIMC];
  int lane = threadIdx.x & 31, wv = threadIdx.x >> 5;
  int ln = lane & 15, hi = lane >> 4;
  int g0 = blockIdx.x * 64 + wv * 16;
  for (int rr = 0; rr < 16; ++rr){
    int g = g0 + rr;
    const float4* xr = (const float4*)(x2 + (size_t)g * DIMC);
    float4 v = xr[lane];
    float s  = v.x + v.y + v.z + v.w;
    float s2 = v.x*v.x + v.y*v.y + v.z*v.z + v.w*v.w;
#pragma unroll
    for (int d = 1; d < 32; d <<= 1){ s += __shfl_xor(s, d, 32); s2 += __shfl_xor(s2, d, 32); }
    float mu  = s  * (1.0f / DIMC);
    float var = s2 * (1.0f / DIMC) - mu * mu;
    float inv = rsqrtf(var + 1e-5f);
    int c0 = lane * 4;
    unsigned short* dst = lnA + (wv * 16 + rr) * DIMC + c0;
    dst[0] = f2bf((v.x - mu) * inv * g2[c0+0] + b2v[c0+0]);
    dst[1] = f2bf((v.y - mu) * inv * g2[c0+1] + b2v[c0+1]);
    dst[2] = f2bf((v.z - mu) * inv * g2[c0+2] + b2v[c0+2]);
    dst[3] = f2bf((v.w - mu) * inv * g2[c0+3] + b2v[c0+3]);
  }
  v16bf af[4];
#pragma unroll
  for (int ks = 0; ks < 4; ++ks) af[ks] = load_a16(lnA + (wv*16)*DIMC + ks*32, DIMC, lane);
  for (int ct = 0; ct < 24; ++ct){
    F8 c;
#pragma unroll
    for (int i = 0; i < 8; ++i) c.f[i] = 0.f;
#pragma unroll
    for (int ks = 0; ks < 4; ++ks)
      c.v = wmma_bf(af[ks], load_bnk(w1 + (ct*16)*DIMC + ks*32, DIMC, lane), c.v);
    int o = ct*16 + ln;
    float bo = bias1[o];
#pragma unroll
    for (int r = 0; r < 8; ++r){
      int g = g0 + r + 8*hi;
      float hv = c.f[r] + bo;
      hv = 0.5f * hv * (1.0f + erff(hv * 0.70710678118f));   // exact GELU
      hbuf[(size_t)g*HIDDEN + o] = f2bf(hv);
    }
  }
}

// ------------------------------------------------- FC2 + bias + residual -> out
__global__ __launch_bounds__(128) void k_fc2(
    const unsigned short* __restrict__ hbuf, const unsigned short* __restrict__ w2,
    const float* __restrict__ bias2, const float* __restrict__ x2, float* __restrict__ out)
{
  int lane = threadIdx.x & 31, wv = threadIdx.x >> 5;
  int ln = lane & 15, hi = lane >> 4;
  int g0 = blockIdx.x * 64 + wv * 16;
  v16bf af[12];
#pragma unroll
  for (int ks = 0; ks < 12; ++ks) af[ks] = load_a16(hbuf + (size_t)g0*HIDDEN + ks*32, HIDDEN, lane);
  for (int ct = 0; ct < 8; ++ct){
    F8 c;
#pragma unroll
    for (int i = 0; i < 8; ++i) c.f[i] = 0.f;
#pragma unroll
    for (int ks = 0; ks < 12; ++ks)
      c.v = wmma_bf(af[ks], load_bnk(w2 + (ct*16)*HIDDEN + ks*32, HIDDEN, lane), c.v);
    int o = ct*16 + ln;
    float bo = bias2[o];
#pragma unroll
    for (int r = 0; r < 8; ++r){
      int g = g0 + r + 8*hi;
      out[(size_t)g*DIMC + o] = c.f[r] + bo + x2[(size_t)g*DIMC + o];
    }
  }
}

// ----------------------------------------------------------------- launcher
extern "C" void kernel_launch(void* const* d_in, const int* in_sizes, int n_in,
                              void* d_out, int out_size, void* d_ws, size_t ws_size,
                              hipStream_t stream) {
  const float* x      = (const float*)d_in[0];
  const float* ln1_g  = (const float*)d_in[1];
  const float* ln1_b  = (const float*)d_in[2];
  const float* qkv_w  = (const float*)d_in[3];
  const float* proj_w = (const float*)d_in[4];
  const float* proj_b = (const float*)d_in[5];
  const float* ln2_g  = (const float*)d_in[6];
  const float* ln2_b  = (const float*)d_in[7];
  const float* w1     = (const float*)d_in[8];
  const float* b1     = (const float*)d_in[9];
  const float* w2     = (const float*)d_in[10];
  const float* b2     = (const float*)d_in[11];
  float* out = (float*)d_out;

  char* ws = (char*)d_ws;
  // byte offsets (all 4B aligned); total ~28 MB
  unsigned short* wqkv  = (unsigned short*)(ws + 0);         // 384*128 bf16
  unsigned short* wproj = (unsigned short*)(ws + 98304);     // 128*128 bf16
  unsigned short* w1b   = (unsigned short*)(ws + 131072);    // 384*128 bf16
  unsigned short* w2b   = (unsigned short*)(ws + 229376);    // 128*384 bf16
  unsigned short* qbuf  = (unsigned short*)(ws + 327680);    // B*HEADS*N*32 bf16
  unsigned short* kbuf  = (unsigned short*)(ws + 3538944);
  unsigned short* vbuf  = (unsigned short*)(ws + 6750208);
  unsigned short* attn  = (unsigned short*)(ws + 9961472);   // MTOT*128 bf16
  float*          x2    = (float*)        (ws + 13172736);   // MTOT*128 f32
  unsigned short* hbuf  = (unsigned short*)(ws + 19595264);  // MTOT*384 bf16

  k_cvt<<<(49152 + 255)/256, 256, 0, stream>>>(qkv_w, wqkv, 49152);
  k_cvt<<<(16384 + 255)/256, 256, 0, stream>>>(proj_w, wproj, 16384);
  k_cvt<<<(49152 + 255)/256, 256, 0, stream>>>(w1, w1b, 49152);
  k_cvt<<<(49152 + 255)/256, 256, 0, stream>>>(w2, w2b, 49152);

  k_ln_qkv_rope<<<MTOT/64, 128, 0, stream>>>(x, ln1_g, ln1_b, wqkv, qbuf, kbuf, vbuf);
  k_attn<<<BD*HEADS*HH, 128, 0, stream>>>(qbuf, kbuf, vbuf, attn);
  k_proj<<<MTOT/64, 128, 0, stream>>>(attn, wproj, proj_b, x, x2);
  k_ln_fc1<<<MTOT/64, 128, 0, stream>>>(x2, ln2_g, ln2_b, w1b, b1, hbuf);
  k_fc2<<<MTOT/64, 128, 0, stream>>>(hbuf, w2b, b2, x2, out);
}
